// Decoder_46394236732042
// MI455X (gfx1250) — compile-verified, matrix operand order checked by reading, not run on hardware
//
#include <hip/hip_runtime.h>
#include <hip/hip_bf16.h>
#include <math.h>

// Problem dims (fixed by the reference)
#define B_  64
#define T_  256
#define S_  512
#define E_  512
#define H_  1024
#define V_  2048
#define L_  6
#define KW_ 3

#define SCALE_ 0.70710678118654752f

typedef __attribute__((ext_vector_type(16))) __bf16 v16bf;
typedef __attribute__((ext_vector_type(8)))  __bf16 v8bf;
typedef __attribute__((ext_vector_type(8)))  float  v8f;

union FragBF { v16bf v; v8bf h[2]; };

// ---------------------------------------------------------------------------
// Generic batched GEMM:  C[z] = A[z] @ Bt[z]^T (+bias), f32 accum via WMMA.
//   A : bf16, row-major [M x K], row stride lda (rows may overlap: conv!)
//   Bt: bf16, row-major [N x K], row stride ldb (i.e. B pre-transposed)
//   C : f32 [M x N], ldc; optional bf16 mirror C16 (same ldc)
// Block tile 128x256, 8 waves (2 M x 4 N), wave tile 64x64, K-step 32.
// 16 WMMAs per 16 global_load_b128 per wave per K-step (1:1 density).
// Fragment loads follow the CDNA5 16-bit A/B layout: lane = (row, half),
// half 0 holds K[0..7]+K[16..23], half 1 holds K[8..15]+K[24..31].
// ---------------------------------------------------------------------------
__global__ __launch_bounds__(256) void gemm_bf16_wmma(
    const __bf16* __restrict__ A,  long strideA, int lda,
    const __bf16* __restrict__ Bt, long strideB, int ldb,
    float* __restrict__ C,  long strideC, int ldc,
    const float* __restrict__ bias,
    __bf16* __restrict__ C16, long strideC16,
    int Kdim)
{
  const int z  = blockIdx.z;
  const __bf16* Ab = A  + (size_t)z * strideA;
  const __bf16* Bb = Bt + (size_t)z * strideB;
  const int m0 = blockIdx.y * 128;
  const int n0 = blockIdx.x * 256;
  const int wave = threadIdx.x >> 5;
  const int lane = threadIdx.x & 31;
  const int wm = (wave >> 2) * 64;   // 2 waves along M (covers 128)
  const int wn = (wave & 3) * 64;    // 4 waves along N (covers 256)
  const int rl = lane & 15;
  const int hf = lane >> 4;

  v8f acc[4][4] = {};

  const __bf16* arow[4];
  const __bf16* brow[4];
#pragma unroll
  for (int i = 0; i < 4; ++i)
    arow[i] = Ab + (size_t)(m0 + wm + 16 * i + rl) * lda + 8 * hf;
#pragma unroll
  for (int j = 0; j < 4; ++j)
    brow[j] = Bb + (size_t)(n0 + wn + 16 * j + rl) * ldb + 8 * hf;

  for (int k0 = 0; k0 < Kdim; k0 += 32) {
    FragBF fa[4], fb[4];
#pragma unroll
    for (int i = 0; i < 4; ++i) {
      fa[i].h[0] = *(const v8bf*)(arow[i] + k0);
      fa[i].h[1] = *(const v8bf*)(arow[i] + k0 + 16);
    }
#pragma unroll
    for (int j = 0; j < 4; ++j) {
      fb[j].h[0] = *(const v8bf*)(brow[j] + k0);
      fb[j].h[1] = *(const v8bf*)(brow[j] + k0 + 16);
    }
#pragma unroll
    for (int i = 0; i < 4; ++i)
#pragma unroll
      for (int j = 0; j < 4; ++j)
        acc[i][j] = __builtin_amdgcn_wmma_f32_16x16x32_bf16(
            false, fa[i].v, false, fb[j].v, (short)0, acc[i][j], false, false);
  }

  float*  Cb   = C + (size_t)z * strideC;
  __bf16* C16b = C16 ? (C16 + (size_t)z * strideC16) : nullptr;
#pragma unroll
  for (int j = 0; j < 4; ++j) {
    const int n  = n0 + wn + 16 * j + rl;
    const float bv = bias ? bias[n] : 0.0f;
#pragma unroll
    for (int i = 0; i < 4; ++i) {
      const int mb = m0 + wm + 16 * i + 8 * hf;
#pragma unroll
      for (int r = 0; r < 8; ++r) {
        const float v = acc[i][j][r] + bv;
        const size_t idx = (size_t)(mb + r) * ldc + n;
        Cb[idx] = v;
        if (C16b) C16b[idx] = (__bf16)v;
      }
    }
  }
}

// ---------------------------------------------------------------------------
// Elementwise / prep kernels
// ---------------------------------------------------------------------------
__global__ void k_embed(const int* __restrict__ trg,
                        const float* __restrict__ tok,
                        const float* __restrict__ pos,
                        float* __restrict__ embF, __bf16* __restrict__ embB)
{
  size_t i = (size_t)blockIdx.x * blockDim.x + threadIdx.x;
  if (i >= (size_t)B_ * T_ * E_) return;
  int e = (int)(i % E_);
  size_t bt = i / E_;
  int t = (int)(bt % T_);
  int b = (int)(bt / T_);
  int token = trg[t * B_ + b];                 // trg is [T, B]
  float v = tok[(size_t)token * E_ + e] + pos[(size_t)t * E_ + e];
  embF[i] = v;
  embB[i] = (__bf16)v;
}

__global__ void k_cvt_bf16(const float* __restrict__ in,
                           __bf16* __restrict__ out, size_t n)
{
  size_t i = (size_t)blockIdx.x * blockDim.x + threadIdx.x;
  if (i < n) out[i] = (__bf16)in[i];
}

// out[z][c][r] = (bf16) in[z][r][c]   (32x32 LDS tiles; R,C multiples of 32)
__global__ void k_transpose_cvt(const float* __restrict__ in,
                                __bf16* __restrict__ out,
                                int R, int Ccols, long inB, long outB)
{
  __shared__ float tile[32][33];
  const int z = blockIdx.z;
  const float* ib = in + (size_t)z * inB;
  __bf16* ob = out + (size_t)z * outB;
  const int c0 = blockIdx.x * 32, r0 = blockIdx.y * 32;
  const int tx = threadIdx.x & 31, ty = threadIdx.x >> 5;   // 32x8
#pragma unroll
  for (int i = 0; i < 32; i += 8)
    tile[ty + i][tx] = ib[(size_t)(r0 + ty + i) * Ccols + c0 + tx];
  __syncthreads();
#pragma unroll
  for (int i = 0; i < 32; i += 8)
    ob[(size_t)(c0 + ty + i) * R + r0 + tx] = (__bf16)tile[tx][ty + i];
}

// convwT[l][o][k*H + i] = conv_w[l][o][i][k]
__global__ void k_pack_convw(const float* __restrict__ w,
                             __bf16* __restrict__ out)
{
  size_t i = (size_t)blockIdx.x * blockDim.x + threadIdx.x;
  const size_t n = (size_t)L_ * 2 * H_ * 3 * H_;
  if (i >= n) return;
  int kk = (int)(i % (3 * H_));
  size_t lo = i / (3 * H_);
  int o = (int)(lo % (2 * H_));
  int l = (int)(lo / (2 * H_));
  int k  = kk / H_;
  int ii = kk % H_;
  out[i] = (__bf16)w[(((size_t)l * 2 * H_ + o) * H_ + ii) * KW_ + k];
}

// x_pad[b, tt, h] = (tt<2) ? 0 : bf16(x[b, tt-2, h])
__global__ void k_pack_pad(const float* __restrict__ x,
                           __bf16* __restrict__ xp)
{
  size_t i = (size_t)blockIdx.x * blockDim.x + threadIdx.x;
  if (i >= (size_t)B_ * (T_ + 2) * H_) return;
  int h = (int)(i % H_);
  size_t bt = i / H_;
  int tt = (int)(bt % (T_ + 2));
  int b  = (int)(bt / (T_ + 2));
  xp[i] = (tt < 2) ? (__bf16)0.0f
                   : (__bf16)x[((size_t)b * T_ + (tt - 2)) * H_ + h];
}

// GLU over channels: out = a * sigmoid(g)
__global__ void k_glu(const float* __restrict__ conv,
                      float* __restrict__ outF, __bf16* __restrict__ outB)
{
  size_t i = (size_t)blockIdx.x * blockDim.x + threadIdx.x;
  if (i >= (size_t)B_ * T_ * H_) return;
  int h = (int)(i % H_);
  size_t bt = i / H_;
  float a = conv[bt * 2 * H_ + h];
  float g = conv[bt * 2 * H_ + H_ + h];
  float v = a / (1.0f + __expf(-g));
  outF[i] = v;
  outB[i] = (__bf16)v;
}

// combined = (conved_emb + embedded) * scale  -> bf16
__global__ void k_combined(const float* __restrict__ ce,
                           const float* __restrict__ emb,
                           __bf16* __restrict__ out)
{
  size_t i = (size_t)blockIdx.x * blockDim.x + threadIdx.x;
  if (i >= (size_t)B_ * T_ * E_) return;
  out[i] = (__bf16)((ce[i] + emb[i]) * SCALE_);
}

// conved2 = (conved + attH)*scale; x = (conved2 + x)*scale; repack x_pad row
__global__ void k_residual(const float* __restrict__ conved,
                           const float* __restrict__ attH,
                           float* __restrict__ x,
                           __bf16* __restrict__ xp)
{
  size_t i = (size_t)blockIdx.x * blockDim.x + threadIdx.x;
  if (i >= (size_t)B_ * T_ * H_) return;
  int h = (int)(i % H_);
  size_t bt = i / H_;
  int t = (int)(bt % T_);
  int b = (int)(bt / T_);
  float c2 = (conved[i] + attH[i]) * SCALE_;
  float xn = (c2 + x[i]) * SCALE_;
  x[i] = xn;
  xp[((size_t)b * (T_ + 2) + t + 2) * H_ + h] = (__bf16)xn;
}

// wave-per-row softmax over S_=512 columns (wave32, shuffle reductions)
__global__ __launch_bounds__(256) void k_softmax(
    const float* __restrict__ energy,
    float* __restrict__ attnF, __bf16* __restrict__ attnB)
{
  const int row  = blockIdx.x * 8 + (threadIdx.x >> 5);
  const int lane = threadIdx.x & 31;
  const float* e = energy + (size_t)row * S_;
  float vals[S_ / 32];
  float mx = -INFINITY;
#pragma unroll
  for (int i = 0; i < S_ / 32; ++i) {
    vals[i] = e[lane + 32 * i];
    mx = fmaxf(mx, vals[i]);
  }
#pragma unroll
  for (int m = 16; m > 0; m >>= 1) mx = fmaxf(mx, __shfl_xor(mx, m, 32));
  float s = 0.0f;
#pragma unroll
  for (int i = 0; i < S_ / 32; ++i) { vals[i] = __expf(vals[i] - mx); s += vals[i]; }
#pragma unroll
  for (int m = 16; m > 0; m >>= 1) s += __shfl_xor(s, m, 32);
  const float inv = 1.0f / s;
#pragma unroll
  for (int i = 0; i < S_ / 32; ++i) {
    float p = vals[i] * inv;
    size_t idx = (size_t)row * S_ + lane + 32 * i;
    attnF[idx] = p;
    attnB[idx] = (__bf16)p;
  }
}

// ---------------------------------------------------------------------------
// Host orchestration
// ---------------------------------------------------------------------------
static inline dim3 flat_grid(size_t n) { return dim3((unsigned)((n + 255) / 256)); }

extern "C" void kernel_launch(void* const* d_in, const int* in_sizes, int n_in,
                              void* d_out, int out_size, void* d_ws, size_t ws_size,
                              hipStream_t stream)
{
  const int*   trg      = (const int*)  d_in[0];
  const float* encConv  = (const float*)d_in[1];   // [B,S,E]
  const float* encComb  = (const float*)d_in[2];   // [B,S,E]
  const float* tok      = (const float*)d_in[3];
  const float* pos      = (const float*)d_in[4];
  const float* w_e2h    = (const float*)d_in[5];
  const float* b_e2h    = (const float*)d_in[6];
  const float* w_h2e    = (const float*)d_in[7];
  const float* b_h2e    = (const float*)d_in[8];
  const float* w_ah2e   = (const float*)d_in[9];
  const float* b_ah2e   = (const float*)d_in[10];
  const float* w_ae2h   = (const float*)d_in[11];
  const float* b_ae2h   = (const float*)d_in[12];
  const float* w_out    = (const float*)d_in[13];
  const float* b_out    = (const float*)d_in[14];
  const float* conv_w   = (const float*)d_in[15];
  const float* conv_b   = (const float*)d_in[16];

  float* outp    = (float*)d_out;                       // [B,T,V]
  float* attnOut = outp + (size_t)B_ * T_ * V_;         // [B,T,S]

  // Workspace carve-up (~590 MB total, buffers reused across pipeline stages)
  char* ws = (char*)d_ws;
  size_t off = 0;
  auto alloc = [&](size_t bytes) -> char* {
    char* p = ws + off;
    off += (bytes + 255) & ~(size_t)255;
    return p;
  };
  float*  embF     = (float*) alloc((size_t)B_*T_*E_*4);        // embedded f32 (live whole call)
  __bf16* bufE16a  = (__bf16*)alloc((size_t)B_*T_*E_*2);        // embedded16 -> combined16 -> attended16
  __bf16* bufE16b  = (__bf16*)alloc((size_t)B_*T_*E_*2);        // attn16 -> out_emb16
  float*  xF       = (float*) alloc((size_t)B_*T_*H_*4);        // residual stream x (f32 master)
  __bf16* xPad     = (__bf16*)alloc((size_t)B_*(T_+2)*H_*2);    // causal-padded x, bf16
  float*  convOut  = (float*) alloc((size_t)B_*T_*2*H_*4);      // conv output [B,T,2H]; attH aliases front half
  float*  attH     = convOut;                                   // [B,T,H] (disjoint in time from convOut use)
  float*  convedF  = (float*) alloc((size_t)B_*T_*H_*4);        // GLU output f32
  __bf16* conved16 = (__bf16*)alloc((size_t)B_*T_*H_*2);        // GLU output bf16
  float*  bufA     = (float*) alloc((size_t)B_*T_*E_*4);        // conved_emb -> energy -> attended -> out_emb
  __bf16* we2hT    = (__bf16*)alloc((size_t)H_*E_*2);           // [H,E]
  __bf16* wh2eT    = (__bf16*)alloc((size_t)E_*H_*2);           // [E,H]
  __bf16* wah2eT   = (__bf16*)alloc((size_t)E_*H_*2);           // [E,H]
  __bf16* wae2hT   = (__bf16*)alloc((size_t)H_*E_*2);           // [H,E]
  __bf16* woutT    = (__bf16*)alloc((size_t)V_*E_*2);           // [V,E]
  __bf16* convwT   = (__bf16*)alloc((size_t)L_*2*H_*3*H_*2);    // [L][2H][3H]
  __bf16* encC16   = (__bf16*)alloc((size_t)B_*S_*E_*2);        // enc_conved bf16 [B,S,E]
  __bf16* encCbT   = (__bf16*)alloc((size_t)B_*E_*S_*2);        // enc_combined^T bf16 [B,E,S]
  (void)ws_size; (void)in_sizes; (void)n_in; (void)out_size;

  // ---- prep: bf16 weight/encoder layouts -------------------------------
  k_cvt_bf16<<<flat_grid((size_t)B_*S_*E_), 256, 0, stream>>>(encConv, encC16, (size_t)B_*S_*E_);
  k_transpose_cvt<<<dim3(E_/32, S_/32, B_), 256, 0, stream>>>(encComb, encCbT, S_, E_, (long)S_*E_, (long)E_*S_);
  k_transpose_cvt<<<dim3(H_/32, E_/32, 1), 256, 0, stream>>>(w_e2h,  we2hT,  E_, H_, 0, 0);
  k_transpose_cvt<<<dim3(E_/32, H_/32, 1), 256, 0, stream>>>(w_h2e,  wh2eT,  H_, E_, 0, 0);
  k_transpose_cvt<<<dim3(E_/32, H_/32, 1), 256, 0, stream>>>(w_ah2e, wah2eT, H_, E_, 0, 0);
  k_transpose_cvt<<<dim3(H_/32, E_/32, 1), 256, 0, stream>>>(w_ae2h, wae2hT, E_, H_, 0, 0);
  k_transpose_cvt<<<dim3(V_/32, E_/32, 1), 256, 0, stream>>>(w_out,  woutT,  E_, V_, 0, 0);
  k_pack_convw<<<flat_grid((size_t)L_*2*H_*3*H_), 256, 0, stream>>>(conv_w, convwT);

  // ---- embedding + e2h --------------------------------------------------
  k_embed<<<flat_grid((size_t)B_*T_*E_), 256, 0, stream>>>(trg, tok, pos, embF, bufE16a);
  gemm_bf16_wmma<<<dim3(H_/256, T_/128, B_), 256, 0, stream>>>(
      bufE16a, (long)T_*E_, E_,  we2hT, 0, E_,
      xF, (long)T_*H_, H_,  b_e2h,  nullptr, 0,  E_);
  k_pack_pad<<<flat_grid((size_t)B_*(T_+2)*H_), 256, 0, stream>>>(xF, xPad);

  // ---- decoder layers ---------------------------------------------------
  for (int l = 0; l < L_; ++l) {
    // causal conv as one GEMM: A rows overlap (lda=H, K=3H)
    gemm_bf16_wmma<<<dim3(2*H_/256, T_/128, B_), 256, 0, stream>>>(
        xPad, (long)(T_+2)*H_, H_,  convwT + (size_t)l*2*H_*3*H_, 0, 3*H_,
        convOut, (long)T_*2*H_, 2*H_,  conv_b + (size_t)l*2*H_,  nullptr, 0,  3*H_);
    k_glu<<<flat_grid((size_t)B_*T_*H_), 256, 0, stream>>>(convOut, convedF, conved16);
    // conved -> emb space
    gemm_bf16_wmma<<<dim3(E_/256, T_/128, B_), 256, 0, stream>>>(
        conved16, (long)T_*H_, H_,  wah2eT, 0, H_,
        bufA, (long)T_*E_, E_,  b_ah2e,  nullptr, 0,  H_);
    k_combined<<<flat_grid((size_t)B_*T_*E_), 256, 0, stream>>>(bufA, embF, bufE16a);
    // energy = combined @ enc_conved^T
    gemm_bf16_wmma<<<dim3(S_/256, T_/128, B_), 256, 0, stream>>>(
        bufE16a, (long)T_*E_, E_,  encC16, (long)S_*E_, E_,
        bufA, (long)T_*S_, S_,  nullptr,  nullptr, 0,  E_);
    k_softmax<<<dim3(B_*T_/8), 256, 0, stream>>>(bufA, attnOut, bufE16b);
    // attended = attn @ enc_combined (fused bf16 mirror for next GEMM)
    gemm_bf16_wmma<<<dim3(E_/256, T_/128, B_), 256, 0, stream>>>(
        bufE16b, (long)T_*S_, S_,  encCbT, (long)E_*S_, S_,
        bufA, (long)T_*E_, E_,  nullptr,  bufE16a, (long)T_*E_,  S_);
    // attended -> hidden space
    gemm_bf16_wmma<<<dim3(H_/256, T_/128, B_), 256, 0, stream>>>(
        bufE16a, (long)T_*E_, E_,  wae2hT, 0, E_,
        attH, (long)T_*H_, H_,  b_ae2h,  nullptr, 0,  E_);
    k_residual<<<flat_grid((size_t)B_*T_*H_), 256, 0, stream>>>(convedF, attH, xF, xPad);
  }

  // ---- output head ------------------------------------------------------
  // A = unpadded rows of xPad (per-batch base shifted by 2 rows)
  gemm_bf16_wmma<<<dim3(E_/256, T_/128, B_), 256, 0, stream>>>(
      xPad + 2*H_, (long)(T_+2)*H_, H_,  wh2eT, 0, H_,
      bufA, (long)T_*E_, E_,  b_h2e,  bufE16b, (long)T_*E_,  H_);
  gemm_bf16_wmma<<<dim3(V_/256, T_/128, B_), 256, 0, stream>>>(
      bufE16b, (long)T_*E_, E_,  woutT, 0, E_,
      outp, (long)T_*V_, V_,  b_out,  nullptr, 0,  E_);
}
